// CLIPCaptioningModel_16621523435608
// MI455X (gfx1250) — matrix operand order repaired and checked by reading, not compiled, not run on hardware
//
#include <hip/hip_runtime.h>
#include <hip/hip_bf16.h>

typedef __attribute__((ext_vector_type(16))) __bf16 v16bf;
typedef __attribute__((ext_vector_type(8)))  __bf16 bf16x8;
typedef __attribute__((ext_vector_type(4)))  __bf16 bf16x4;
typedef __attribute__((ext_vector_type(2)))  __bf16 bf16x2;
typedef __attribute__((ext_vector_type(8)))  float  v8f;

#define TILE_M 128
#define TILE_N 128
#define TILE_K 32
#define LDS_PITCH 40   // elements; 80B row stride: 16B-aligned, spreads banks

// Generic C = act(A @ B + bias) (+ residual)
// A: [M,K] f32 row-major. B: f32, bmode=0 -> [K,N]; bmode=1 -> head-blocked [N/64][K][64]
// REQUIRES: N % 128 == 0, K % 32 == 0 (true for all call sites).
// act: 0=none, 1=exact gelu. residual may be nullptr; bias may be nullptr.
__global__ __launch_bounds__(256)
void gemm_wmma_bf16(const float* __restrict__ A, const float* __restrict__ B,
                    const float* __restrict__ bias, const float* __restrict__ residual,
                    float* __restrict__ C, int M, int N, int K, int bmode, int act)
{
    __shared__ __bf16 As[TILE_M][LDS_PITCH];   // 10 KB, rows m, k-contiguous
    __shared__ __bf16 Bt[TILE_N][LDS_PITCH];   // 10 KB, rows n, k-contiguous (transposed)

    const int tid   = threadIdx.x;
    const int lane  = tid & 31;
    const int wave  = tid >> 5;       // 0..7
    const int waveM = wave >> 1;      // 0..3  -> 32-row strip
    const int waveN = wave & 1;       // 0..1  -> 64-col strip
    const int blockRow = blockIdx.y * TILE_M;
    const int blockCol = blockIdx.x * TILE_N;

    v8f acc[2][4] = {};

    const int half = lane >> 4;
    const int l16  = lane & 15;

    // ---- per-thread staging geometry (hoisted out of K loop) ----
    // A: 4 float4 units
    int    aR[4], aKq[4];
    size_t aBase[4];
    #pragma unroll
    for (int i = 0; i < 4; ++i) {
        int idx4 = i * 256 + tid;
        aR[i]  = idx4 >> 3;                 // 0..127
        aKq[i] = (idx4 & 7) * 4;            // 0..28
        int gr  = blockRow + aR[i];
        int grc = (gr < M) ? gr : (M - 1);  // branch-free clamp
        aBase[i] = (size_t)grc * K + aKq[i];
    }
    // B: 2 units of (k-pair, 4-col group); two k-rows per unit
    int    bN[2], bK[2];
    size_t bBase[2];
    const size_t bStride = bmode ? 64 : (size_t)N;   // offset step per +1 in k
    #pragma unroll
    for (int i = 0; i < 2; ++i) {
        int u  = i * 256 + tid;
        int ng = u & 31;
        int kp = u >> 5;
        bN[i] = ng * 4;
        bK[i] = kp * 2;
        int gc = blockCol + bN[i];          // always < N (N % 128 == 0)
        if (bmode) bBase[i] = ((size_t)(gc >> 6) * K + bK[i]) * 64 + (gc & 63);
        else       bBase[i] = (size_t)bK[i] * N + gc;
    }

    float4 ra[4], rb0[2], rb1[2];

    // ---- fetch tile k0=0 into registers ----
    #pragma unroll
    for (int i = 0; i < 4; ++i) ra[i] = *(const float4*)(A + aBase[i]);
    #pragma unroll
    for (int i = 0; i < 2; ++i) {
        rb0[i] = *(const float4*)(B + bBase[i]);
        rb1[i] = *(const float4*)(B + bBase[i] + bStride);
    }

    for (int k0 = 0; k0 < K; k0 += TILE_K) {
        // ---- regs -> LDS (bf16) ----
        #pragma unroll
        for (int i = 0; i < 4; ++i) {
            bf16x4 p;
            p[0] = (__bf16)ra[i].x; p[1] = (__bf16)ra[i].y;
            p[2] = (__bf16)ra[i].z; p[3] = (__bf16)ra[i].w;
            *(bf16x4*)(&As[aR[i]][aKq[i]]) = p;
        }
        #pragma unroll
        for (int i = 0; i < 2; ++i) {
            const float r0a[4] = {rb0[i].x, rb0[i].y, rb0[i].z, rb0[i].w};
            const float r1a[4] = {rb1[i].x, rb1[i].y, rb1[i].z, rb1[i].w};
            #pragma unroll
            for (int j = 0; j < 4; ++j) {
                bf16x2 p;
                p[0] = (__bf16)r0a[j];
                p[1] = (__bf16)r1a[j];
                *(bf16x2*)(&Bt[bN[i] + j][bK[i]]) = p;
            }
        }
        __syncthreads();

        // ---- prefetch next tile into registers (overlaps with WMMA below) ----
        const int kn = k0 + TILE_K;
        if (kn < K) {
            #pragma unroll
            for (int i = 0; i < 4; ++i)
                ra[i] = *(const float4*)(A + aBase[i] + kn);
            #pragma unroll
            for (int i = 0; i < 2; ++i) {
                size_t o = bBase[i] + (size_t)kn * bStride;
                rb0[i] = *(const float4*)(B + o);
                rb1[i] = *(const float4*)(B + o + bStride);
            }
        }

        // ---- fragments: contiguous 16B LDS reads + concat ----
        v16bf aF[2], bF[4];
        #pragma unroll
        for (int sm = 0; sm < 2; ++sm) {
            const __bf16* arow = &As[waveM * 32 + sm * 16 + l16][0];
            bf16x8 a0 = *(const bf16x8*)(arow + half * 8);
            bf16x8 a1 = *(const bf16x8*)(arow + 16 + half * 8);
            aF[sm] = __builtin_shufflevector(a0, a1, 0,1,2,3,4,5,6,7,8,9,10,11,12,13,14,15);
        }
        #pragma unroll
        for (int sn = 0; sn < 4; ++sn) {
            const __bf16* brow = &Bt[waveN * 64 + sn * 16 + l16][0];
            bf16x8 b0 = *(const bf16x8*)(brow + half * 8);
            bf16x8 b1 = *(const bf16x8*)(brow + 16 + half * 8);
            bF[sn] = __builtin_shufflevector(b0, b1, 0,1,2,3,4,5,6,7,8,9,10,11,12,13,14,15);
        }

        // ---- 8 x v_wmma_f32_16x16x32_bf16 ----
        #pragma unroll
        for (int sm = 0; sm < 2; ++sm)
            #pragma unroll
            for (int sn = 0; sn < 4; ++sn)
                acc[sm][sn] = __builtin_amdgcn_wmma_f32_16x16x32_bf16(
                    false, aF[sm], false, bF[sn], (short)0, acc[sm][sn], false, false);
        __syncthreads();
    }

    // ---- epilogue ----
    const int mOff = half * 8;
    #pragma unroll
    for (int sm = 0; sm < 2; ++sm) {
        #pragma unroll
        for (int sn = 0; sn < 4; ++sn) {
            int col = blockCol + waveN * 64 + sn * 16 + l16;   // always < N
            float bv = bias ? bias[col] : 0.0f;
            #pragma unroll
            for (int r = 0; r < 8; ++r) {
                int row = blockRow + waveM * 32 + sm * 16 + mOff + r;
                if (row >= M) continue;
                float val = acc[sm][sn][r] + bv;
                if (act == 1)
                    val = 0.5f * val * (1.0f + erff(val * 0.70710678118654752440f));
                if (residual) val += residual[(size_t)row * N + col];
                C[(size_t)row * N + col] = val;
            }
        }
    }
}

__global__ __launch_bounds__(256)
void embed_k(const int* __restrict__ tokens, const float* __restrict__ tok_emb,
             const float* __restrict__ pos_emb, float* __restrict__ x, int T_, int E_)
{
    int row = blockIdx.x;             // b*T + t
    int t = row % T_;
    int tok = tokens[row];
    const float* te = tok_emb + (size_t)tok * E_;
    const float* pe = pos_emb + (size_t)t * E_;
    float* xr = x + (size_t)row * E_;
    for (int e = threadIdx.x; e < E_; e += blockDim.x)
        xr[e] = te[e] + pe[e];
}

__global__ __launch_bounds__(256)
void layernorm_k(const float* __restrict__ x, const float* __restrict__ g,
                 const float* __restrict__ b, float* __restrict__ y, int E_)
{
    int row = blockIdx.x;
    const float* xr = x + (size_t)row * E_;
    float* yr = y + (size_t)row * E_;
    __shared__ float red[256];

    float s = 0.0f;
    for (int e = threadIdx.x; e < E_; e += 256) s += xr[e];
    red[threadIdx.x] = s; __syncthreads();
    for (int off = 128; off > 0; off >>= 1) {
        if (threadIdx.x < off) red[threadIdx.x] += red[threadIdx.x + off];
        __syncthreads();
    }
    float mu = red[0] / (float)E_;
    __syncthreads();

    float v = 0.0f;
    for (int e = threadIdx.x; e < E_; e += 256) { float d = xr[e] - mu; v += d * d; }
    red[threadIdx.x] = v; __syncthreads();
    for (int off = 128; off > 0; off >>= 1) {
        if (threadIdx.x < off) red[threadIdx.x] += red[threadIdx.x + off];
        __syncthreads();
    }
    float rstd = rsqrtf(red[0] / (float)E_ + 1e-5f);

    for (int e = threadIdx.x; e < 512; e += 256)
        yr[e] = (xr[e] - mu) * rstd * g[e] + b[e];
}

// One block (64 threads) per (b,h,t) row. q/out: [B*Tq, H*HS]; k,v: [B*S, H*HS]
__global__ __launch_bounds__(64)
void attn_k(const float* __restrict__ q, const float* __restrict__ k,
            const float* __restrict__ v, float* __restrict__ out,
            int Tq, int S, int Hn, int HSn, int causal)
{
    int idx = blockIdx.x;
    int t = idx % Tq;
    int h = (idx / Tq) % Hn;
    int b = idx / (Tq * Hn);
    int EH = Hn * HSn;

    const float* qr = q + (size_t)(b * Tq + t) * EH + h * HSn;
    __shared__ float sc[160];
    __shared__ float red[64];

    int lim = causal ? (t + 1) : S;
    float scale = rsqrtf((float)HSn);

    for (int s = threadIdx.x; s < lim; s += 64) {
        const float* kr = k + (size_t)(b * S + s) * EH + h * HSn;
        float d = 0.0f;
        for (int e = 0; e < HSn; ++e) d += qr[e] * kr[e];
        sc[s] = d * scale;
    }
    __syncthreads();

    float m = -1e30f;
    for (int s = threadIdx.x; s < lim; s += 64) m = fmaxf(m, sc[s]);
    red[threadIdx.x] = m; __syncthreads();
    for (int off = 32; off > 0; off >>= 1) {
        if (threadIdx.x < off) red[threadIdx.x] = fmaxf(red[threadIdx.x], red[threadIdx.x + off]);
        __syncthreads();
    }
    m = red[0]; __syncthreads();

    float ssum = 0.0f;
    for (int s = threadIdx.x; s < lim; s += 64) {
        float e2 = expf(sc[s] - m);
        sc[s] = e2;
        ssum += e2;
    }
    red[threadIdx.x] = ssum; __syncthreads();
    for (int off = 32; off > 0; off >>= 1) {
        if (threadIdx.x < off) red[threadIdx.x] += red[threadIdx.x + off];
        __syncthreads();
    }
    float inv = 1.0f / red[0];
    __syncthreads();

    int e = threadIdx.x;              // HS == 64 == blockDim.x
    float o = 0.0f;
    for (int s = 0; s < lim; ++s)
        o += sc[s] * v[(size_t)(b * S + s) * EH + h * HSn + e];
    out[(size_t)(b * Tq + t) * EH + h * HSn + e] = o * inv;
}

static inline dim3 gemm_grid(int M, int N) {
    return dim3((N + TILE_N - 1) / TILE_N, (M + TILE_M - 1) / TILE_M);
}

extern "C" void kernel_launch(void* const* d_in, const int* in_sizes, int n_in,
                              void* d_out, int out_size, void* d_ws, size_t ws_size,
                              hipStream_t stream) {
    (void)in_sizes; (void)n_in; (void)out_size; (void)ws_size;

    constexpr int B_ = 16, T_ = 128, SI = 50, E_ = 512, H_ = 8, HS_ = 64, L_ = 6, V_ = 32000, D_ = 768;
    constexpr int BT = B_ * T_;       // 2048
    constexpr int BS = B_ * SI;       // 800
    constexpr int EH = H_ * HS_;      // 512
    constexpr int FF = 4 * E_;        // 2048

    int ii = 0;
    const int*   tokens  = (const int*)  d_in[ii++];
    const float* img     = (const float*)d_in[ii++];
    const float* Wvp     = (const float*)d_in[ii++];
    const float* bvp     = (const float*)d_in[ii++];
    const float* tok_emb = (const float*)d_in[ii++];
    const float* pos_emb = (const float*)d_in[ii++];
    const float* ln1_g   = (const float*)d_in[ii++];
    const float* ln1_b   = (const float*)d_in[ii++];
    const float* Wq1     = (const float*)d_in[ii++];
    const float* Wk1     = (const float*)d_in[ii++];
    const float* Wv1     = (const float*)d_in[ii++];
    const float* Wo1     = (const float*)d_in[ii++];
    const float* bo1     = (const float*)d_in[ii++];
    const float* ln2_g   = (const float*)d_in[ii++];
    const float* ln2_b   = (const float*)d_in[ii++];
    const float* Wq2     = (const float*)d_in[ii++];
    const float* Wk2     = (const float*)d_in[ii++];
    const float* Wv2     = (const float*)d_in[ii++];
    const float* Wo2     = (const float*)d_in[ii++];
    const float* bo2     = (const float*)d_in[ii++];
    const float* ln3_g   = (const float*)d_in[ii++];
    const float* ln3_b   = (const float*)d_in[ii++];
    const float* W1      = (const float*)d_in[ii++];
    const float* b1      = (const float*)d_in[ii++];
    const float* W2      = (const float*)d_in[ii++];
    const float* b2      = (const float*)d_in[ii++];
    const float* lnf_g   = (const float*)d_in[ii++];
    const float* lnf_b   = (const float*)d_in[ii++];
    const float* W_lm    = (const float*)d_in[ii++];
    const float* b_lm    = (const float*)d_in[ii++];

    const size_t sW   = (size_t)H_ * E_ * HS_;   // 262144
    const size_t sWo  = (size_t)EH * E_;         // 262144
    const size_t sW1  = (size_t)E_ * FF;         // 1048576
    const size_t sW2  = (size_t)FF * E_;         // 1048576

    float* ws = (float*)d_ws;
    size_t off = 0;
    float* x    = ws + off; off += (size_t)BT * E_;
    float* xln  = ws + off; off += (size_t)BT * E_;
    float* enc  = ws + off; off += (size_t)BS * E_;
    float* q    = ws + off; off += (size_t)BT * EH;
    float* kbuf = ws + off; off += (size_t)BT * EH;
    float* vbuf = ws + off; off += (size_t)BT * EH;
    float* att  = ws + off; off += (size_t)BT * EH;
    float* mlp  = ws + off; off += (size_t)BT * FF;

    // visual projection: enc = img @ Wvp + bvp   [800,768]x[768,512]
    gemm_wmma_bf16<<<gemm_grid(BS, E_), 256, 0, stream>>>(
        img, Wvp, bvp, nullptr, enc, BS, E_, D_, 0, 0);

    // token + positional embedding
    embed_k<<<BT, 256, 0, stream>>>(tokens, tok_emb, pos_emb, x, T_, E_);

    for (int l = 0; l < L_; ++l) {
        // ---------- self-attention ----------
        layernorm_k<<<BT, 256, 0, stream>>>(x, ln1_g + (size_t)l * E_, ln1_b + (size_t)l * E_, xln, E_);
        gemm_wmma_bf16<<<gemm_grid(BT, EH), 256, 0, stream>>>(
            xln, Wq1 + (size_t)l * sW, nullptr, nullptr, q, BT, EH, E_, 1, 0);
        gemm_wmma_bf16<<<gemm_grid(BT, EH), 256, 0, stream>>>(
            xln, Wk1 + (size_t)l * sW, nullptr, nullptr, kbuf, BT, EH, E_, 1, 0);
        gemm_wmma_bf16<<<gemm_grid(BT, EH), 256, 0, stream>>>(
            xln, Wv1 + (size_t)l * sW, nullptr, nullptr, vbuf, BT, EH, E_, 1, 0);
        attn_k<<<B_ * H_ * T_, 64, 0, stream>>>(q, kbuf, vbuf, att, T_, T_, H_, HS_, 1);
        gemm_wmma_bf16<<<gemm_grid(BT, E_), 256, 0, stream>>>(
            att, Wo1 + (size_t)l * sWo, bo1 + (size_t)l * E_, x, x, BT, E_, EH, 0, 0);

        // ---------- cross-attention ----------
        layernorm_k<<<BT, 256, 0, stream>>>(x, ln2_g + (size_t)l * E_, ln2_b + (size_t)l * E_, xln, E_);
        gemm_wmma_bf16<<<gemm_grid(BT, EH), 256, 0, stream>>>(
            xln, Wq2 + (size_t)l * sW, nullptr, nullptr, q, BT, EH, E_, 1, 0);
        gemm_wmma_bf16<<<gemm_grid(BS, EH), 256, 0, stream>>>(
            enc, Wk2 + (size_t)l * sW, nullptr, nullptr, kbuf, BS, EH, E_, 1, 0);
        gemm_wmma_bf16<<<gemm_grid(BS, EH), 256, 0, stream>>>(
            enc, Wv2 + (size_t)l * sW, nullptr, nullptr, vbuf, BS, EH, E_, 1, 0);
        attn_k<<<B_ * H_ * T_, 64, 0, stream>>>(q, kbuf, vbuf, att, T_, SI, H_, HS_, 0);
        gemm_wmma_bf16<<<gemm_grid(BT, E_), 256, 0, stream>>>(
            att, Wo2 + (size_t)l * sWo, bo2 + (size_t)l * E_, x, x, BT, E_, EH, 0, 0);

        // ---------- MLP ----------
        layernorm_k<<<BT, 256, 0, stream>>>(x, ln3_g + (size_t)l * E_, ln3_b + (size_t)l * E_, xln, E_);
        gemm_wmma_bf16<<<gemm_grid(BT, FF), 256, 0, stream>>>(
            xln, W1 + (size_t)l * sW1, b1 + (size_t)l * FF, nullptr, mlp, BT, FF, E_, 0, 1);
        gemm_wmma_bf16<<<gemm_grid(BT, E_), 256, 0, stream>>>(
            mlp, W2 + (size_t)l * sW2, b2 + (size_t)l * E_, x, x, BT, E_, FF, 0, 0);
    }

    // final LN + LM head -> logits [2048, 32000]
    layernorm_k<<<BT, 256, 0, stream>>>(x, lnf_g, lnf_b, xln, E_);
    gemm_wmma_bf16<<<gemm_grid(BT, V_), 256, 0, stream>>>(
        xln, W_lm, b_lm, nullptr, (float*)d_out, BT, V_, E_, 0, 0);
}
